// LKAAttention_22814866276922
// MI455X (gfx1250) — matrix operand order repaired and checked by reading, not compiled, not run on hardware
//
#include <hip/hip_runtime.h>
#include <hip/hip_bf16.h>
#include <math.h>

typedef __attribute__((ext_vector_type(16))) __bf16 v16bf;
typedef __attribute__((ext_vector_type(8)))  __bf16 v8bf;
typedef __attribute__((ext_vector_type(8)))  float  v8f;

#define SCALE 0.17677669529663687f  // 1/sqrt(32)

__device__ __forceinline__ v16bf cat16(v8bf lo, v8bf hi) {
  return __builtin_shufflevector(lo, hi, 0, 1, 2, 3, 4, 5, 6, 7, 8, 9, 10, 11,
                                 12, 13, 14, 15);
}

__device__ __forceinline__ v8f wmma_bf16(v16bf a, v16bf b, v8f c) {
  return __builtin_amdgcn_wmma_f32_16x16x32_bf16(false, a, false, b, (short)0, c,
                                                 false, false);
}

__device__ __forceinline__ float phi_scale(float v) {
  return (v > 0.0f ? v + 1.0f : __expf(v)) * SCALE;  // (elu+1)/sqrt(D)
}

// ---------------------------------------------------------------------------
// Tiled GEMM: epi(A[M,K] @ W[K,N] + bias). Block tile 128x64, 8 waves,
// wave tile 32x32 (2x2 WMMA accums), K step 32.
// Double-buffered LDS + register-staged pipeline: loads for tile t+1 are in
// flight while tile t is computed; one barrier per iteration.
// As stored [m][k] with k contiguous so each A fragment is one
// contiguous 32B v16bf LDS read; Bs stored [n][k] so B fragments are too.
// EPI: 0 = phi -> bf16 row-major (Q)
//      1 = phi -> bf16 transposed per-head [bh][d][token] (K)
//      2 = plain -> bf16 transposed per-head (V)
//      3 = plain -> f32 row-major (final output)
// ---------------------------------------------------------------------------
template <int EPI, bool ABF16>
__global__ __launch_bounds__(256) void gemm_k(const void* __restrict__ Aptr,
                                              const float* __restrict__ W,
                                              const float* __restrict__ bias,
                                              void* __restrict__ Cptr,
                                              int M, int N, int K) {
  __shared__ __align__(32) __bf16 As[2][128][64];  // [buf][m][k]
  __shared__ __align__(32) __bf16 Bs[2][64][64];   // [buf][n][k]

  const int tid  = threadIdx.x;
  const int row0 = blockIdx.x * 128;
  const int n0   = blockIdx.y * 64;
  const int wave = tid >> 5, lane = tid & 31;
  const int wm = (wave & 3) * 32;
  const int wn = (wave >> 2) * 32;
  const int m16 = lane & 15, hh = lane >> 4;

  v8f acc[2][2];
#pragma unroll
  for (int mi = 0; mi < 2; ++mi)
#pragma unroll
    for (int ni = 0; ni < 2; ++ni)
#pragma unroll
      for (int i = 0; i < 8; ++i) acc[mi][ni][i] = 0.0f;

  const int ar = tid >> 1, ac = (tid & 1) * 16;   // A stage: 2 thr/row
  const int bkr = tid >> 3, bnb = (tid & 7) * 8;  // W stage: 8 thr/row
  const int ad0 = ac, ad1 = ac + 8;  // contiguous k dests

  // register staging for the pipeline
  float4 a0, a1, a2, a3;  // fp32 A path
  v8bf h0, h1;            // bf16 A path
  float4 w0, w1;          // weights

  // ---- prologue: load + stage tile 0 into buffer 0 ----
  if (ABF16) {
    const __bf16* Ab = (const __bf16*)Aptr + (size_t)(row0 + ar) * K + ac;
    h0 = *(const v8bf*)Ab;
    h1 = *(const v8bf*)(Ab + 8);
  } else {
    const float4* Af = (const float4*)((const float*)Aptr +
                                       (size_t)(row0 + ar) * K + ac);
    a0 = Af[0]; a1 = Af[1]; a2 = Af[2]; a3 = Af[3];
  }
  {
    const float4* Wp =
        (const float4*)(W + (size_t)bkr * N + n0 + bnb);
    w0 = Wp[0]; w1 = Wp[1];
  }
  if (ABF16) {
    *(v8bf*)&As[0][ar][ad0] = h0;
    *(v8bf*)&As[0][ar][ad1] = h1;
  } else {
    v8bf p0, p1;
    p0[0] = (__bf16)a0.x; p0[1] = (__bf16)a0.y; p0[2] = (__bf16)a0.z;
    p0[3] = (__bf16)a0.w; p0[4] = (__bf16)a1.x; p0[5] = (__bf16)a1.y;
    p0[6] = (__bf16)a1.z; p0[7] = (__bf16)a1.w;
    p1[0] = (__bf16)a2.x; p1[1] = (__bf16)a2.y; p1[2] = (__bf16)a2.z;
    p1[3] = (__bf16)a2.w; p1[4] = (__bf16)a3.x; p1[5] = (__bf16)a3.y;
    p1[6] = (__bf16)a3.z; p1[7] = (__bf16)a3.w;
    *(v8bf*)&As[0][ar][ad0] = p0;
    *(v8bf*)&As[0][ar][ad1] = p1;
  }
  Bs[0][bnb + 0][bkr] = (__bf16)w0.x;
  Bs[0][bnb + 1][bkr] = (__bf16)w0.y;
  Bs[0][bnb + 2][bkr] = (__bf16)w0.z;
  Bs[0][bnb + 3][bkr] = (__bf16)w0.w;
  Bs[0][bnb + 4][bkr] = (__bf16)w1.x;
  Bs[0][bnb + 5][bkr] = (__bf16)w1.y;
  Bs[0][bnb + 6][bkr] = (__bf16)w1.z;
  Bs[0][bnb + 7][bkr] = (__bf16)w1.w;
  __syncthreads();

#pragma unroll 2
  for (int kt = 0; kt < K; kt += 32) {
    const int buf = (kt >> 5) & 1;
    const bool more = (kt + 32) < K;

    // ---- issue global loads for tile t+1 (stay in flight over compute) ----
    if (more) {
      if (ABF16) {
        const __bf16* Ab =
            (const __bf16*)Aptr + (size_t)(row0 + ar) * K + kt + 32 + ac;
        h0 = *(const v8bf*)Ab;
        h1 = *(const v8bf*)(Ab + 8);
      } else {
        const float4* Af = (const float4*)((const float*)Aptr +
                                           (size_t)(row0 + ar) * K + kt + 32 + ac);
        a0 = Af[0]; a1 = Af[1]; a2 = Af[2]; a3 = Af[3];
      }
      const float4* Wp =
          (const float4*)(W + (size_t)(kt + 32 + bkr) * N + n0 + bnb);
      w0 = Wp[0]; w1 = Wp[1];
    }

    // ---- compute tile t from LDS[buf] ----
    v16bf afr[2], bfr[2];
#pragma unroll
    for (int mi = 0; mi < 2; ++mi)
      afr[mi] = *(const v16bf*)&As[buf][wm + mi * 16 + m16][hh * 16];
#pragma unroll
    for (int ni = 0; ni < 2; ++ni)
      bfr[ni] = *(const v16bf*)&Bs[buf][wn + ni * 16 + m16][hh * 16];
#pragma unroll
    for (int mi = 0; mi < 2; ++mi)
#pragma unroll
      for (int ni = 0; ni < 2; ++ni)
        acc[mi][ni] = wmma_bf16(afr[mi], bfr[ni], acc[mi][ni]);

    // ---- convert + store staged tile t+1 into LDS[buf^1] ----
    if (more) {
      const int nb = buf ^ 1;
      if (ABF16) {
        *(v8bf*)&As[nb][ar][ad0] = h0;
        *(v8bf*)&As[nb][ar][ad1] = h1;
      } else {
        v8bf p0, p1;
        p0[0] = (__bf16)a0.x; p0[1] = (__bf16)a0.y; p0[2] = (__bf16)a0.z;
        p0[3] = (__bf16)a0.w; p0[4] = (__bf16)a1.x; p0[5] = (__bf16)a1.y;
        p0[6] = (__bf16)a1.z; p0[7] = (__bf16)a1.w;
        p1[0] = (__bf16)a2.x; p1[1] = (__bf16)a2.y; p1[2] = (__bf16)a2.z;
        p1[3] = (__bf16)a2.w; p1[4] = (__bf16)a3.x; p1[5] = (__bf16)a3.y;
        p1[6] = (__bf16)a3.z; p1[7] = (__bf16)a3.w;
        *(v8bf*)&As[nb][ar][ad0] = p0;
        *(v8bf*)&As[nb][ar][ad1] = p1;
      }
      Bs[nb][bnb + 0][bkr] = (__bf16)w0.x;
      Bs[nb][bnb + 1][bkr] = (__bf16)w0.y;
      Bs[nb][bnb + 2][bkr] = (__bf16)w0.z;
      Bs[nb][bnb + 3][bkr] = (__bf16)w0.w;
      Bs[nb][bnb + 4][bkr] = (__bf16)w1.x;
      Bs[nb][bnb + 5][bkr] = (__bf16)w1.y;
      Bs[nb][bnb + 6][bkr] = (__bf16)w1.z;
      Bs[nb][bnb + 7][bkr] = (__bf16)w1.w;
      __syncthreads();
    }
  }

  // ---- epilogue ----
#pragma unroll
  for (int mi = 0; mi < 2; ++mi)
#pragma unroll
    for (int ni = 0; ni < 2; ++ni) {
      const int c = n0 + wn + ni * 16 + m16;
      const float bb = bias[c];
      if (EPI == 1 || EPI == 2) {
        // packed transposed store: 8 consecutive tokens per lane -> one v8bf
        const int r0 = row0 + wm + mi * 16 + 8 * hh;
        v8bf pk;
#pragma unroll
        for (int i = 0; i < 8; ++i) {
          float v = acc[mi][ni][i] + bb;
          if (EPI == 1) v = phi_scale(v);
          pk[i] = (__bf16)v;
        }
        const int bh = (r0 >> 13) * 16 + (c >> 5);  // b*16 + h
        __bf16* dst = (__bf16*)Cptr +
                      ((size_t)(bh * 32 + (c & 31))) * 8192 + (r0 & 8191);
        *(v8bf*)dst = pk;
      } else {
#pragma unroll
        for (int i = 0; i < 8; ++i) {
          const int r = row0 + wm + mi * 16 + i + 8 * hh;
          float v = acc[mi][ni][i] + bb;
          if (EPI == 0) v = phi_scale(v);
          if (EPI == 3) {
            ((float*)Cptr)[(size_t)r * N + c] = v;
          } else {
            ((__bf16*)Cptr)[(size_t)r * N + c] = (__bf16)v;
          }
        }
      }
    }
}

// ---------------------------------------------------------------------------
// kv = K^T V (32x32 per head) + ksum via WMMA vs all-ones B.
// KT/VT are [bh][d][token] bf16 -> all fragment loads are contiguous.
// grid = (16 chunks of 512 tokens, 64 heads), 1 wave per block.
// ---------------------------------------------------------------------------
__global__ __launch_bounds__(32) void kv_reduce(const __bf16* __restrict__ KT,
                                                const __bf16* __restrict__ VT,
                                                float* __restrict__ kvbuf,
                                                float* __restrict__ ksum) {
  const int bh   = blockIdx.y;
  const int s0   = blockIdx.x * 512;
  const int lane = threadIdx.x;
  const int m16 = lane & 15, hh = lane >> 4;

  v8f acc[2][2], ks[2];
#pragma unroll
  for (int di = 0; di < 2; ++di) {
#pragma unroll
    for (int i = 0; i < 8; ++i) ks[di][i] = 0.0f;
#pragma unroll
    for (int ei = 0; ei < 2; ++ei)
#pragma unroll
      for (int i = 0; i < 8; ++i) acc[di][ei][i] = 0.0f;
  }
  v16bf ones;
#pragma unroll
  for (int j = 0; j < 16; ++j) ones[j] = (__bf16)1.0f;

  const __bf16* krow[2];
  const __bf16* vrow[2];
#pragma unroll
  for (int t = 0; t < 2; ++t) {
    krow[t] = KT + (size_t)(bh * 32 + t * 16 + m16) * 8192 + s0;
    vrow[t] = VT + (size_t)(bh * 32 + t * 16 + m16) * 8192 + s0;
  }

  for (int s = 0; s < 512; s += 32) {
    v16bf afr[2], bfr[2];
#pragma unroll
    for (int di = 0; di < 2; ++di)
      afr[di] = *(const v16bf*)(krow[di] + s + hh * 16);
#pragma unroll
    for (int ei = 0; ei < 2; ++ei)
      bfr[ei] = *(const v16bf*)(vrow[ei] + s + hh * 16);
#pragma unroll
    for (int di = 0; di < 2; ++di) {
      ks[di] = wmma_bf16(afr[di], ones, ks[di]);  // ksum partials
#pragma unroll
      for (int ei = 0; ei < 2; ++ei)
        acc[di][ei] = wmma_bf16(afr[di], bfr[ei], acc[di][ei]);
    }
  }

  float* kvh = kvbuf + bh * 1024;
#pragma unroll
  for (int di = 0; di < 2; ++di) {
#pragma unroll
    for (int ei = 0; ei < 2; ++ei)
#pragma unroll
      for (int i = 0; i < 8; ++i) {
        const int d = di * 16 + i + 8 * hh;
        const int e = ei * 16 + m16;
        atomicAdd(&kvh[d * 32 + e], acc[di][ei][i]);
      }
    if (m16 == 0) {
#pragma unroll
      for (int i = 0; i < 8; ++i)
        atomicAdd(&ksum[bh * 32 + di * 16 + i + 8 * hh], ks[di][i]);
    }
  }
}

// kv f32 -> bf16 in B-fragment layout [bh][e][d]; ksum f32 -> bf16
__global__ __launch_bounds__(32) void kv_finalize(const float* __restrict__ kvbuf,
                                                  const float* __restrict__ ksum,
                                                  __bf16* __restrict__ kvT,
                                                  __bf16* __restrict__ ksumb) {
  const int bh = blockIdx.x;
  const int e = threadIdx.x;
  const float* kvh = kvbuf + bh * 1024;
  __bf16* o = kvT + (size_t)(bh * 32 + e) * 32;
#pragma unroll
  for (int d = 0; d < 32; ++d) o[d] = (__bf16)kvh[d * 32 + e];
  ksumb[bh * 32 + e] = (__bf16)ksum[bh * 32 + e];
}

// ---------------------------------------------------------------------------
// num = Q @ kv (one WMMA per 16x16 tile), den = Q @ broadcast(ksum) via WMMA
// (den lands in the same accumulator layout as num -> lane-local divide),
// attn = num * rcp(den) in bf16. grid = (128 groups of 64 rows, 64 heads).
// ---------------------------------------------------------------------------
__global__ __launch_bounds__(32) void num_den(const __bf16* __restrict__ Qb,
                                              const __bf16* __restrict__ kvT,
                                              const __bf16* __restrict__ ksumb,
                                              __bf16* __restrict__ attn) {
  const int bh   = blockIdx.y;
  const int col0 = (bh & 15) * 32;
  const int row0 = (bh >> 4) * 8192 + blockIdx.x * 64;
  const int lane = threadIdx.x;
  const int m16 = lane & 15, hh = lane >> 4;

  // kv B fragments: contiguous v16bf loads from transposed bf16 tile
  v16bf bfr[2];
#pragma unroll
  for (int ei = 0; ei < 2; ++ei)
    bfr[ei] =
        *(const v16bf*)(kvT + (size_t)(bh * 32 + ei * 16 + m16) * 32 + hh * 16);
  // ksum broadcast B matrix: every column equals ksum (halves j -> k=hh*16+j)
  const v16bf bden = *(const v16bf*)(ksumb + bh * 32 + hh * 16);

#pragma unroll
  for (int mt = 0; mt < 4; ++mt) {
    const __bf16* ap = Qb + (size_t)(row0 + mt * 16 + m16) * 512 + col0;
    v16bf afr = *(const v16bf*)(ap + hh * 16);

    v8f dacc;
#pragma unroll
    for (int i = 0; i < 8; ++i) dacc[i] = 0.0f;
    dacc = wmma_bf16(afr, bden, dacc);  // den[m] replicated across columns
    v8f rden;
#pragma unroll
    for (int i = 0; i < 8; ++i) rden[i] = __builtin_amdgcn_rcpf(dacc[i]);

#pragma unroll
    for (int ei = 0; ei < 2; ++ei) {
      v8f acc;
#pragma unroll
      for (int i = 0; i < 8; ++i) acc[i] = 0.0f;
      acc = wmma_bf16(afr, bfr[ei], acc);
#pragma unroll
      for (int i = 0; i < 8; ++i) {
        const int mloc = mt * 16 + i + 8 * hh;
        attn[(size_t)(row0 + mloc) * 512 + col0 + ei * 16 + m16] =
            (__bf16)(acc[i] * rden[i]);
      }
    }
  }
}

__global__ void zero_f32(float* p, int n) {
  const int i = blockIdx.x * blockDim.x + threadIdx.x;
  if (i < n) p[i] = 0.0f;
}

extern "C" void kernel_launch(void* const* d_in, const int* in_sizes, int n_in,
                              void* d_out, int out_size, void* d_ws, size_t ws_size,
                              hipStream_t stream) {
  const float* x  = (const float*)d_in[0];
  const float* Wq = (const float*)d_in[1];
  const float* bq = (const float*)d_in[2];
  const float* Wk = (const float*)d_in[3];
  const float* bk = (const float*)d_in[4];
  const float* Wv = (const float*)d_in[5];
  const float* bv = (const float*)d_in[6];
  const float* Wo = (const float*)d_in[7];
  const float* bo = (const float*)d_in[8];
  float* out = (float*)d_out;

  const int M = 32768, N = 512, K = 512;
  __bf16* Qb   = (__bf16*)d_ws;              // [32768][512] row-major
  __bf16* KT   = Qb + (size_t)M * N;         // [64 bh][32 d][8192 tok]
  __bf16* VT   = KT + (size_t)M * N;         // [64 bh][32 e][8192 tok]
  __bf16* attn = VT + (size_t)M * N;         // [32768][512] row-major
  float* kvbuf = (float*)(attn + (size_t)M * N);  // [64][32][32] f32
  float* ksum  = kvbuf + 64 * 1024;               // [64][32] f32
  __bf16* kvT  = (__bf16*)(ksum + 64 * 32);       // [64][32 e][32 d] bf16
  __bf16* ksumb = kvT + 64 * 32 * 32;             // [64][32] bf16
  const int nzero = 64 * 1024 + 64 * 32;

  zero_f32<<<(nzero + 255) / 256, 256, 0, stream>>>(kvbuf, nzero);

  dim3 gg(M / 128, N / 64);
  gemm_k<0, false><<<gg, 256, 0, stream>>>(x, Wq, bq, Qb, M, N, K);  // phi(Q)
  gemm_k<1, false><<<gg, 256, 0, stream>>>(x, Wk, bk, KT, M, N, K);  // phi(K)^T
  gemm_k<2, false><<<gg, 256, 0, stream>>>(x, Wv, bv, VT, M, N, K);  // V^T

  kv_reduce<<<dim3(16, 64), 32, 0, stream>>>(KT, VT, kvbuf, ksum);
  kv_finalize<<<64, 32, 0, stream>>>(kvbuf, ksum, kvT, ksumb);
  num_den<<<dim3(128, 64), 32, 0, stream>>>(Qb, kvT, ksumb, attn);

  gemm_k<3, true><<<gg, 256, 0, stream>>>(attn, Wo, bo, out, M, N, K);  // f32 out
}
